// TransConvLayer_46772193854042
// MI455X (gfx1250) — matrix operand order repaired and checked by reading, not compiled, not run on hardware
//
#include <hip/hip_runtime.h>

// Lorentz (hyperbolic) multi-head attention for MI455X / gfx1250.
// B=4, S=2048, IN=512, H=8, OUT=64, D=OUT+1=65.
// bf16 WMMA (v_wmma_f32_16x16x32_bf16) for all GEMM stages, f32 accumulation,
// flash-attention fused softmax. K/V tiles are double-buffered in LDS and
// moved by the Tensor Data Mover (tensor_load_to_lds + s_wait_tensorcnt),
// with a cooperative vector-load fallback if the TDM builtin is unavailable.

#define B_N   4
#define S_LEN 2048
#define IN_C  512
#define H_N   8
#define OUT_C 64
#define DQ    96   // padded Lorentz dim for Q/K (65 -> 96, 3 chunks of K=32)
#define DV    80   // padded Lorentz dim for V / ctx (65 -> 80, 5 tiles of N=16)

#if __has_builtin(__builtin_amdgcn_tensor_load_to_lds) && \
    __has_builtin(__builtin_amdgcn_s_wait_tensorcnt)
#define USE_TDM 1
#else
#define USE_TDM 0
#endif

typedef __attribute__((ext_vector_type(16))) __bf16 v16bf;
typedef __attribute__((ext_vector_type(8)))  __bf16 v8bf;
typedef __attribute__((ext_vector_type(4)))  __bf16 v4bf;
typedef __attribute__((ext_vector_type(8)))  float  v8f;
typedef __attribute__((ext_vector_type(4)))  unsigned int v4u;
typedef __attribute__((ext_vector_type(8)))  int v8i;
typedef __attribute__((ext_vector_type(4)))  int v4i;

static __device__ inline v16bf ld16(const __bf16* p0, const __bf16* p1) {
  union { v16bf v; v8bf h[2]; } u;
  u.h[0] = *(const v8bf*)p0;
  u.h[1] = *(const v8bf*)p1;
  return u.v;
}

static __device__ inline v8f wmma_bf16(v16bf a, v16bf b, v8f c) {
  // emits v_wmma_f32_16x16x32_bf16
  return __builtin_amdgcn_wmma_f32_16x16x32_bf16(false, a, false, b,
                                                 (short)0, c, false, false);
}

#if USE_TDM
// Issue a TDM 2-D tile load: global (row-major, stride0 elements of bf16)
// -> LDS at byte offset lds_off, tile = tile_d1 rows x tile_d0 elements.
// 6-arg builtin form (clang-23 / therock-10.0): extra int32x8 group, zeroed
// for 2-D tensors.
static __device__ inline void tdm_load_2d(unsigned int lds_off,
                                          const __bf16* gptr,
                                          unsigned int tile_d0,
                                          unsigned int tile_d1,
                                          unsigned int stride0) {
  unsigned long long ga = (unsigned long long)(uintptr_t)gptr;
  v4u g0;
  g0[0] = 1u;                                   // count=1, user descriptor
  g0[1] = lds_off;                              // lds_addr (bytes)
  g0[2] = (unsigned int)ga;                     // global_addr[31:0]
  g0[3] = (unsigned int)((ga >> 32) & 0x01FFFFFFu) | (2u << 30); // [56:32]|type=2
  v8i g1;
  g1[0] = (int)(1u << 16);                      // data_size=1 (2 bytes)
  g1[1] = (int)(tile_d0 << 16);                 // tensor_dim0[15:0] @63:48
  g1[2] = (int)(tile_d1 << 16);                 // tensor_dim1[15:0] @95:80
  g1[3] = (int)(tile_d0 << 16);                 // tile_dim0 @127:112
  g1[4] = (int)tile_d1;                         // tile_dim1 @143:128
  g1[5] = (int)stride0;                         // tensor_dim0_stride[31:0]
  g1[6] = 0;
  g1[7] = 0;
  v4i z4 = {0, 0, 0, 0};                        // groups 2/3: 2-D tensor
  v8i z8 = {0, 0, 0, 0, 0, 0, 0, 0};            // unused extra group
  __builtin_amdgcn_tensor_load_to_lds(g0, g1, z4, z4, z8, 0);
}
#endif

// ---------------------------------------------------------------- convert ---
__global__ void cvt_f32_bf16(const float* __restrict__ in,
                             __bf16* __restrict__ out, int n) {
  int i = (blockIdx.x * 256 + threadIdx.x) * 4;
  if (i < n) {
    float4 f = *(const float4*)(in + i);
    v4bf o = {(__bf16)f.x, (__bf16)f.y, (__bf16)f.z, (__bf16)f.w};
    *(v4bf*)(out + i) = o;
  }
}

// ------------------------------------------------------------- projection ---
// sp = x @ W^T + b per (proj, head); Lorentz time t = sqrt(|sp|^2 + 1).
// proj 0 -> Q [B,H,S,96] with col0 = -t (Lorentz sign folded)
// proj 1 -> K [B,H,S,96] with col0 = +t
// proj 2 -> Vt [B,H,80,S] transposed (row0 = +t) for P·V B-fragments.
__global__ __launch_bounds__(256) void proj_kernel(
    const __bf16* __restrict__ xq, const __bf16* __restrict__ xs,
    const __bf16* __restrict__ Wall,
    const float* __restrict__ bq, const float* __restrict__ bk,
    const float* __restrict__ bv,
    __bf16* __restrict__ Qb, __bf16* __restrict__ Kb,
    __bf16* __restrict__ Vt) {
  const int wave = threadIdx.x >> 5;
  const int lane = threadIdx.x & 31;
  const int half = lane >> 4;
  const int l16  = lane & 15;
  const int zz   = blockIdx.z;          // proj*H + h
  const int proj = zz >> 3;
  const int h    = zz & 7;

  const __bf16* x = (proj == 0) ? xq : xs;
  const __bf16* W = Wall + (size_t)zz * (OUT_C * IN_C);
  const float* bp = (proj == 0) ? bq : (proj == 1) ? bk : bv;

  const int rflat0 = blockIdx.x * 128 + wave * 16;     // 16 rows per wave
  const __bf16* aptr = x + (size_t)(rflat0 + l16) * IN_C;

  v8f zero = {0.f,0.f,0.f,0.f,0.f,0.f,0.f,0.f};
  v8f acc[4] = {zero, zero, zero, zero};

  for (int kk = 0; kk < IN_C; kk += 32) {
    const int akb = kk + half * 8;                     // A frag chunks
    v16bf a = ld16(aptr + akb, aptr + akb + 16);
#pragma unroll
    for (int n = 0; n < 4; ++n) {
      const __bf16* wp = W + (size_t)(n * 16 + l16) * IN_C + kk + half * 8;
      v16bf bfrag = ld16(wp, wp + 16);
      acc[n] = wmma_bf16(a, bfrag, acc[n]);
    }
  }

  // + bias (b[h][out]); Lorentz time per row
#pragma unroll
  for (int n = 0; n < 4; ++n) {
    float bval = bp[h * OUT_C + n * 16 + l16];
#pragma unroll
    for (int r = 0; r < 8; ++r) acc[n][r] += bval;
  }

  float t[8];
#pragma unroll
  for (int r = 0; r < 8; ++r) {
    float v = 0.f;
#pragma unroll
    for (int n = 0; n < 4; ++n) v += acc[n][r] * acc[n][r];
    v += __shfl_xor(v, 1, 32);
    v += __shfl_xor(v, 2, 32);
    v += __shfl_xor(v, 4, 32);
    v += __shfl_xor(v, 8, 32);
    t[r] = sqrtf(v + 1.0f);                            // curvature K = 1
  }

#pragma unroll
  for (int r = 0; r < 8; ++r) {
    const int row = rflat0 + r + half * 8;
    const int b   = row >> 11;                          // S = 2048
    const int s   = row & 2047;
    const size_t bh = (size_t)(b * H_N + h);
    if (proj < 2) {
      __bf16* dst = (proj == 0 ? Qb : Kb) + (bh * S_LEN + s) * DQ;
#pragma unroll
      for (int n = 0; n < 4; ++n)
        dst[1 + n * 16 + l16] = (__bf16)acc[n][r];
      if (l16 == 0)
        dst[0] = (proj == 0) ? (__bf16)(-t[r]) : (__bf16)(t[r]);
    } else {
      __bf16* dst = Vt + bh * DV * S_LEN;
#pragma unroll
      for (int n = 0; n < 4; ++n)
        dst[(size_t)(1 + n * 16 + l16) * S_LEN + s] = (__bf16)acc[n][r];
      if (l16 == 0) dst[s] = (__bf16)t[r];
    }
  }
}

// --------------------------------------------------------- flash attention ---
// Per (b,h): att = (2 + 2*<q,k>_L)/scale + bias, causal softmax, ctx = att@V.
// Block = 8 waves x 16 query rows. 32-key K/V tiles double-buffered in LDS,
// moved by TDM (wave 0 issues, s_wait_tensorcnt, barriers publish to block).
// LDS byte map (static LDS block starts at offset 0):
//   [    0, 12288)  K tiles  2 x 32x96  bf16 (6144 B each)
//   [12288, 22528)  Vt tiles 2 x 80x32  bf16 (5120 B each)
//   [22528, 30720)  per-wave P transpose scratch 8 x 512 bf16
#define K_TILE_B 6144
#define V_TILE_B 5120
#define V_OFF_B  12288
#define P_OFF_B  22528
#define SMEM_B   30720

__global__ __launch_bounds__(256) void attn_kernel(
    const __bf16* __restrict__ Qb, const __bf16* __restrict__ Kb,
    const __bf16* __restrict__ Vt, const float* __restrict__ scale_p,
    const float* __restrict__ bias_p, float* __restrict__ ctx) {
  __shared__ __attribute__((aligned(16))) unsigned char smem[SMEM_B];

  const int tid  = threadIdx.x;
  const int wave = tid >> 5;
  const int lane = tid & 31;
  const int half = lane >> 4;
  const int l16  = lane & 15;
  const int h = blockIdx.y, b = blockIdx.z;
  const int Q0 = blockIdx.x * 128;
  const int q0 = Q0 + wave * 16;
  const size_t bh = (size_t)(b * H_N + h);

  // Q A-fragments, loaded once (3 chunks of K=32 over padded D=96)
  const __bf16* ap = Qb + (bh * S_LEN + q0 + l16) * DQ;
  v16bf qa[3];
#pragma unroll
  for (int c = 0; c < 3; ++c) {
    int kb2 = c * 32 + half * 8;
    qa[c] = ld16(ap + kb2, ap + kb2 + 16);
  }

  const float inv_scale = 1.0f / scale_p[0];
  const float abias = bias_p[0];
  const __bf16* Kbase = Kb + bh * S_LEN * DQ;
  const __bf16* Vbase = Vt + bh * DV * S_LEN;
  __bf16* myp = (__bf16*)(smem + P_OFF_B) + wave * 512;

  v8f zero = {0.f,0.f,0.f,0.f,0.f,0.f,0.f,0.f};
  v8f acco[5] = {zero, zero, zero, zero, zero};
  float m[8], l[8];
#pragma unroll
  for (int r = 0; r < 8; ++r) { m[r] = -1e30f; l[r] = 0.f; }

  const int kend_blk = Q0 + 128;                       // uniform per block
  const int kend_w   = q0 + 16;                        // this wave's causal end

#if USE_TDM
  // prologue: DMA tile 0 into buffer 0 (2 tensor ops: K then V)
  if (wave == 0) {
    tdm_load_2d(0u, Kbase, DQ, 32, DQ);
    tdm_load_2d((unsigned)V_OFF_B, Vbase, 32, DV, S_LEN);
  }
#endif

  for (int kb = 0; kb < kend_blk; kb += 32) {
    const int cur = (kb >> 5) & 1;
    const __bf16* kt = (const __bf16*)(smem + cur * K_TILE_B);
    const __bf16* vt = (const __bf16*)(smem + V_OFF_B + cur * V_TILE_B);

#if USE_TDM
    if (wave == 0) {
      const int kn = kb + 32;
      if (kn < kend_blk) {                             // DMA next tile pair
        tdm_load_2d((unsigned)((cur ^ 1) * K_TILE_B),
                    Kbase + (size_t)kn * DQ, DQ, 32, DQ);
        tdm_load_2d((unsigned)(V_OFF_B + (cur ^ 1) * V_TILE_B),
                    Vbase + kn, 32, DV, S_LEN);
        __builtin_amdgcn_s_wait_tensorcnt(2);          // current pair done
      } else {
        __builtin_amdgcn_s_wait_tensorcnt(0);
      }
    }
    __syncthreads();                                   // tile visible to all
#else
    __syncthreads();                                   // previous readers done
    for (int idx = tid; idx < 384 + 320; idx += 256) { // 16B chunks
      if (idx < 384) {                                 // K: 12 chunks per key
        int key = idx / 12, c8 = idx % 12;
        *(v8bf*)((__bf16*)(smem + cur * K_TILE_B) + key * DQ + c8 * 8) =
            *(const v8bf*)(Kbase + (size_t)(kb + key) * DQ + c8 * 8);
      } else {                                         // V: 4 chunks per dim
        int i2 = idx - 384;
        int d = i2 >> 2, c8 = i2 & 3;
        *(v8bf*)((__bf16*)(smem + V_OFF_B + cur * V_TILE_B) + d * 32 + c8 * 8) =
            *(const v8bf*)(Vbase + (size_t)d * S_LEN + kb + c8 * 8);
      }
    }
    if (kb + 32 < kend_blk) {                          // cache prefetch hints
      __builtin_prefetch(Kbase + (size_t)(kb + 32 + (tid & 31)) * DQ, 0, 0);
      if (tid < DV)
        __builtin_prefetch(Vbase + (size_t)tid * S_LEN + kb + 32, 0, 0);
    }
    __syncthreads();                                   // tile visible to all
#endif

    if (kb < kend_w) {                                 // wave-uniform causal gate
      // ---- QK^T scores: 6 WMMAs from LDS fragments ----
      v8f sc[2];
#pragma unroll
      for (int nt = 0; nt < 2; ++nt) {
        v8f s = zero;
        const __bf16* kp = kt + (nt * 16 + l16) * DQ + half * 8;
#pragma unroll
        for (int c = 0; c < 3; ++c)
          s = wmma_bf16(qa[c], ld16(kp + c * 32, kp + c * 32 + 16), s);
        const int col = kb + nt * 16 + l16;
#pragma unroll
        for (int r = 0; r < 8; ++r) {
          float v = (2.0f + 2.0f * s[r]) * inv_scale + abias;
          const int row = q0 + r + half * 8;
          s[r] = (col <= row) ? v : -1e30f;            // causal mask
        }
        sc[nt] = s;
      }

      // ---- online softmax (row stats across 16 lanes of each half) ----
      float corr[8];
#pragma unroll
      for (int r = 0; r < 8; ++r) {
        float bm = fmaxf(sc[0][r], sc[1][r]);
        bm = fmaxf(bm, __shfl_xor(bm, 1, 32));
        bm = fmaxf(bm, __shfl_xor(bm, 2, 32));
        bm = fmaxf(bm, __shfl_xor(bm, 4, 32));
        bm = fmaxf(bm, __shfl_xor(bm, 8, 32));
        float mn = fmaxf(m[r], bm);
        float p0 = __expf(sc[0][r] - mn);
        float p1 = __expf(sc[1][r] - mn);
        float cr = __expf(m[r] - mn);
        float rs = p0 + p1;
        rs += __shfl_xor(rs, 1, 32);
        rs += __shfl_xor(rs, 2, 32);
        rs += __shfl_xor(rs, 4, 32);
        rs += __shfl_xor(rs, 8, 32);
        l[r] = l[r] * cr + rs;
        m[r] = mn;
        corr[r] = cr;
        sc[0][r] = p0; sc[1][r] = p1;
      }
#pragma unroll
      for (int n = 0; n < 5; ++n)
#pragma unroll
        for (int r = 0; r < 8; ++r) acco[n][r] *= corr[r];

      // ---- transpose P (C-layout -> A-fragment) via per-wave LDS scratch ----
#pragma unroll
      for (int nt = 0; nt < 2; ++nt)
#pragma unroll
        for (int r = 0; r < 8; ++r)
          myp[(r + half * 8) * 32 + nt * 16 + l16] = (__bf16)sc[nt][r];

      const __bf16* pr = myp + l16 * 32 + half * 8;
      v16bf pa = ld16(pr, pr + 16);

      // ---- P @ V: 5 WMMAs from shared Vt tile ----
#pragma unroll
      for (int n = 0; n < 5; ++n) {
        const __bf16* vp = vt + (n * 16 + l16) * 32 + half * 8;
        acco[n] = wmma_bf16(pa, ld16(vp, vp + 16), acco[n]);
      }
    }
    __syncthreads();       // readers done before buffer 'cur' is re-filled
  }

  // normalize by softmax denominator, store ctx [B,H,S,80] f32
#pragma unroll
  for (int r = 0; r < 8; ++r) {
    float inv = 1.0f / l[r];
    const size_t row = bh * S_LEN + (size_t)(q0 + r + half * 8);
#pragma unroll
    for (int n = 0; n < 5; ++n)
      ctx[row * DV + n * 16 + l16] = acco[n][r] * inv;
  }
}

// --------------------------------------------------- head mean + mid point ---
// ave = mean_h ctx; inner = -ave0^2 + sum(ave_s^2); out = ave/sqrt(max(|inner|,1e-8))
__global__ void midpoint_kernel(const float* __restrict__ ctx,
                                float* __restrict__ out) {
  const int t  = threadIdx.x;          // 32 threads = 1 wave per token
  const int rs = blockIdx.x;           // b*S + s
  const int b = rs >> 11, s = rs & 2047;
  float a0 = 0.f, a1 = 0.f, a2 = 0.f;
  for (int h = 0; h < H_N; ++h) {
    const float* p = ctx + (((size_t)(b * H_N + h)) * S_LEN + s) * DV;
    a0 += p[t];
    a1 += p[t + 32];
    if (t == 0) a2 += p[64];
  }
  a0 *= 0.125f; a1 *= 0.125f; a2 *= 0.125f;
  float sq = (t == 0 ? -a0 * a0 : a0 * a0) + a1 * a1 + a2 * a2;
  sq += __shfl_xor(sq, 1, 32);
  sq += __shfl_xor(sq, 2, 32);
  sq += __shfl_xor(sq, 4, 32);
  sq += __shfl_xor(sq, 8, 32);
  sq += __shfl_xor(sq, 16, 32);
  float invd = 1.0f / sqrtf(fmaxf(fabsf(sq), 1e-8f));
  float* o = out + (size_t)rs * 65;
  o[t] = a0 * invd;
  o[t + 32] = a1 * invd;
  if (t == 0) o[64] = a2 * invd;
}

// ------------------------------------------------------------------ launch ---
extern "C" void kernel_launch(void* const* d_in, const int* in_sizes, int n_in,
                              void* d_out, int out_size, void* d_ws,
                              size_t ws_size, hipStream_t stream) {
  const float* query  = (const float*)d_in[0];
  const float* source = (const float*)d_in[1];
  const float* Wq = (const float*)d_in[2];
  const float* bq = (const float*)d_in[3];
  const float* Wk = (const float*)d_in[4];
  const float* bk = (const float*)d_in[5];
  const float* Wv = (const float*)d_in[6];
  const float* bv = (const float*)d_in[7];
  const float* scale = (const float*)d_in[8];
  const float* bias  = (const float*)d_in[9];
  float* out = (float*)d_out;
  char* ws = (char*)d_ws;

  const size_t nX = (size_t)B_N * S_LEN * IN_C;       // 4,194,304
  const size_t nW = (size_t)H_N * OUT_C * IN_C;       // 262,144 per proj
  const size_t nQ = (size_t)B_N * H_N * S_LEN * DQ;   // 6,291,456
  const size_t nV = (size_t)B_N * H_N * DV * S_LEN;   // 5,242,880

  const size_t o_xq  = 0;
  const size_t o_xs  = o_xq + nX * 2;
  const size_t o_w   = o_xs + nX * 2;
  const size_t o_q   = o_w + 3 * nW * 2;
  const size_t o_k   = o_q + nQ * 2;
  const size_t o_vt  = o_k + nQ * 2;
  const size_t o_ctx = o_vt + nV * 2;

  __bf16* xq  = (__bf16*)(ws + o_xq);
  __bf16* xs  = (__bf16*)(ws + o_xs);
  __bf16* wbf = (__bf16*)(ws + o_w);
  __bf16* Qb  = (__bf16*)(ws + o_q);
  __bf16* Kb  = (__bf16*)(ws + o_k);
  __bf16* Vt  = (__bf16*)(ws + o_vt);
  float*  ctx = (float*)(ws + o_ctx);

  // zero padded Q/K/Vt region (buffers arrive poisoned; pads must be 0)
  (void)hipMemsetAsync(ws + o_q, 0, nQ * 2 * 2 + nV * 2, stream);

  cvt_f32_bf16<<<(int)((nX / 4 + 255) / 256), 256, 0, stream>>>(query, xq, (int)nX);
  cvt_f32_bf16<<<(int)((nX / 4 + 255) / 256), 256, 0, stream>>>(source, xs, (int)nX);
  cvt_f32_bf16<<<(int)((nW / 4 + 255) / 256), 256, 0, stream>>>(Wq, wbf, (int)nW);
  cvt_f32_bf16<<<(int)((nW / 4 + 255) / 256), 256, 0, stream>>>(Wk, wbf + nW, (int)nW);
  cvt_f32_bf16<<<(int)((nW / 4 + 255) / 256), 256, 0, stream>>>(Wv, wbf + 2 * nW, (int)nW);

  // 3 projections x 8 heads on grid.z; 128 rows per block over B*S
  proj_kernel<<<dim3((B_N * S_LEN) / 128, 1, 3 * H_N), 256, 0, stream>>>(
      xq, xs, wbf, bq, bk, bv, Qb, Kb, Vt);

  attn_kernel<<<dim3(S_LEN / 128, H_N, B_N), 256, 0, stream>>>(
      Qb, Kb, Vt, scale, bias, ctx);

  midpoint_kernel<<<dim3(B_N * S_LEN), 32, 0, stream>>>(ctx, out);
}
